// LinearAttention_335007449984
// MI455X (gfx1250) — matrix operand order repaired and checked by reading, not compiled, not run on hardware
//
#include <hip/hip_runtime.h>

// ---------------------------------------------------------------------------
// Linear attention for MI455X (gfx1250, wave32, WMMA + async global->LDS).
// B=8, C=512, N=4096, M=64.  All GEMMs via v_wmma_f32_16x16x32_bf16.
// ---------------------------------------------------------------------------

#define BATCH 8
#define CH    512
#define NPOS  4096
#define MQ    64
#define RTOT  (MQ + MQ + CH)   // 640 rows of fused QKV weight
#define EPSV  1e-6f

typedef __attribute__((ext_vector_type(16))) __bf16        v16bf;
typedef __attribute__((ext_vector_type(8)))  float         v8f;
typedef __attribute__((ext_vector_type(4)))  unsigned int  u32x4;

typedef __attribute__((address_space(3))) unsigned short lds_ushort;

union Frag16 { v16bf v; u32x4 q[2]; };

__device__ __forceinline__ unsigned short f2bf(float f) {
  union { float f; unsigned u; } v; v.f = f;
  unsigned u = v.u;
  return (unsigned short)((u + 0x7FFFu + ((u >> 16) & 1u)) >> 16);  // RNE
}
__device__ __forceinline__ float bf2f(unsigned short s) {
  union { unsigned u; float f; } v; v.u = ((unsigned)s) << 16;
  return v.f;
}

// A-fragment: 16x32 bf16, rows stored row-major with contiguous K.
// p = rowBase + kbase + 16*hi  (elements i = K+i, one 32B run)
__device__ __forceinline__ v16bf loadA(const unsigned short* p) {
  Frag16 f;
  f.q[0] = *(const u32x4*)(p);
  f.q[1] = *(const u32x4*)(p + 8);
  return f.v;
}
// B-fragment: 32x16 bf16, memory [col][k] contiguous K.
// p = colBase + kbase + 16*hi  (elements i = K+i, one 32B run)
__device__ __forceinline__ v16bf loadB(const unsigned short* p) {
  Frag16 f;
  f.q[0] = *(const u32x4*)(p);
  f.q[1] = *(const u32x4*)(p + 8);
  return f.v;
}

__device__ __forceinline__ v8f wmma_bf16(v16bf a, v16bf b, v8f c) {
  return __builtin_amdgcn_wmma_f32_16x16x32_bf16(false, a, false, b,
                                                 (short)0, c, false, false);
}

// Async global->LDS 16B copy (per-lane), ASYNCcnt-tracked.
__device__ __forceinline__ void async_copy16(const unsigned short* g,
                                             unsigned short* l) {
  asm volatile("global_load_async_to_lds_b128 %0, %1, off"
               :: "v"((lds_ushort*)l), "v"(g)
               : "memory");
}
__device__ __forceinline__ void wait_async0() {
  asm volatile("s_wait_asynccnt 0x0" ::: "memory");
}

// ---------------------------------------------------------------------------
// K0: transpose+convert x[b][c][n] f32 -> xT[b][n][c] bf16 (LDS-tiled 32x32)
// ---------------------------------------------------------------------------
__global__ void k_transpose(const float* __restrict__ x,
                            unsigned short* __restrict__ xT) {
  __shared__ float tile[32][33];
  const int b  = blockIdx.z;
  const int n0 = blockIdx.x * 32, c0 = blockIdx.y * 32;
  const int tx = threadIdx.x, ty = threadIdx.y;
  const float* xb = x + (size_t)b * CH * NPOS;
#pragma unroll
  for (int i = 0; i < 4; ++i) {
    int c = c0 + ty + i * 8;
    tile[ty + i * 8][tx] = xb[(size_t)c * NPOS + n0 + tx];
  }
  __syncthreads();
  unsigned short* xTb = xT + (size_t)b * NPOS * CH;
#pragma unroll
  for (int i = 0; i < 4; ++i) {
    int n = n0 + ty + i * 8;
    xTb[(size_t)n * CH + c0 + tx] = f2bf(tile[tx][ty + i * 8]);
  }
}

// K0b: pack Wq/Wk/Wv (f32) -> Wcat bf16 [640][512]
__global__ void k_pack_w(const float* __restrict__ Wq, const float* __restrict__ Wk,
                         const float* __restrict__ Wv, unsigned short* __restrict__ Wcat) {
  int idx = blockIdx.x * blockDim.x + threadIdx.x;   // 640*512
  int r = idx >> 9, c = idx & 511;
  float v;
  if (r < MQ)            v = Wq[r * CH + c];
  else if (r < 2 * MQ)   v = Wk[(r - MQ) * CH + c];
  else                   v = Wv[(r - 2 * MQ) * CH + c];
  Wcat[idx] = f2bf(v);
}

// K0c: Ksum[b][m] = EPS (accumulated later by norm kernel)
__global__ void k_init_ksum(float* __restrict__ Ksum) {
  int i = blockIdx.x * blockDim.x + threadIdx.x;
  if (i < BATCH * MQ) Ksum[i] = EPSV;
}

// ---------------------------------------------------------------------------
// K1: fused QKV GEMM.  Out[b][n][r] = sum_c xT[n][c] * Wcat[r][c] + bias[r]
// Block = 8 waves, 128 n-rows x 64 r-cols.  The 64x512 B panel (64 KB,
// contiguous in Wcat) is staged in LDS via async global->LDS b128 copies;
// A-fragments stream from global, software-pipelined in registers.
// grid (N/128=32, RTOT/64=10, B)
// ---------------------------------------------------------------------------
__global__ void __launch_bounds__(256)
k_gemm_qkv(const unsigned short* __restrict__ xT, const unsigned short* __restrict__ Wcat,
           const float* __restrict__ bq, const float* __restrict__ bk,
           const float* __restrict__ bv,
           float* __restrict__ Qf, float* __restrict__ Kf,
           unsigned short* __restrict__ Vc) {
  __shared__ unsigned short ldsB[64 * CH];            // 64 KB
  const int t    = threadIdx.x;
  const int b    = blockIdx.z;
  const int w    = t >> 5;
  const int lane = t & 31, l15 = lane & 15, hi = lane >> 4;
  const int n0   = blockIdx.x * 128 + w * 16;
  const int r0   = blockIdx.y * 64;

  // Stage B panel: rows [r0, r0+64) x k [0,512)  == contiguous 64 KB.
  {
    const unsigned short* g = Wcat + (size_t)r0 * CH;
#pragma unroll
    for (int i = 0; i < 16; ++i) {
      const unsigned idx = (unsigned)(t + 256 * i) * 8;   // 16B chunks
      async_copy16(g + idx, ldsB + idx);
    }
    wait_async0();
  }
  __syncthreads();

  const unsigned short* Arow = xT + (size_t)b * NPOS * CH + (size_t)(n0 + l15) * CH;
  v8f acc[4] = {};
  v16bf af = loadA(Arow + hi * 16);
  for (int kk = 0; kk < CH; kk += 32) {
    v16bf afn;
    if (kk + 32 < CH) {
      __builtin_prefetch(Arow + kk + 160, 0, 1);
      afn = loadA(Arow + kk + 32 + hi * 16);
    }
#pragma unroll
    for (int a = 0; a < 4; ++a) {
      const unsigned short* Bp = ldsB + (size_t)(a * 16 + l15) * CH + kk + hi * 16;
      acc[a] = wmma_bf16(af, loadB(Bp), acc[a]);      // B frag: two ds_load_b128
    }
    if (kk + 32 < CH) af = afn;
  }

  if (r0 == 0) {                       // Q rows 0..63
    float* Qb = Qf + (size_t)b * NPOS * MQ;
#pragma unroll
    for (int a = 0; a < 4; ++a) {
      int m = a * 16 + l15; float bias = bq[m];
#pragma unroll
      for (int rr = 0; rr < 8; ++rr)
        Qb[(size_t)(n0 + rr + 8 * hi) * MQ + m] = acc[a][rr] + bias;
    }
  } else if (r0 == 64) {               // K rows 64..127
    float* Kb = Kf + (size_t)b * NPOS * MQ;
#pragma unroll
    for (int a = 0; a < 4; ++a) {
      int m = a * 16 + l15; float bias = bk[m];
#pragma unroll
      for (int rr = 0; rr < 8; ++rr)
        Kb[(size_t)(n0 + rr + 8 * hi) * MQ + m] = acc[a][rr] + bias;
    }
  } else {                             // V rows 128..639 -> Vc[b][c][n] bf16
#pragma unroll
    for (int a = 0; a < 4; ++a) {
      int c = r0 - 128 + a * 16 + l15; float bias = bv[c];
      unsigned short* Vrow = Vc + ((size_t)b * CH + c) * NPOS;
#pragma unroll
      for (int rr = 0; rr < 8; ++rr)
        Vrow[n0 + rr + 8 * hi] = f2bf(acc[a][rr] + bias);
    }
  }
}

// ---------------------------------------------------------------------------
// K2: per-position L2 norms. Qn[b][n][m] bf16, Kn[b][m][n] bf16,
//     Ksum[b][m] += sum_n Kn (LDS float atomics -> global atomics).
// grid (N/256=16, B)
// ---------------------------------------------------------------------------
__global__ void k_norm(const float* __restrict__ Qf, const float* __restrict__ Kf,
                       unsigned short* __restrict__ Qn, unsigned short* __restrict__ Kn,
                       float* __restrict__ Ksum) {
  __shared__ float sK[MQ];
  const int t = threadIdx.x, b = blockIdx.y;
  if (t < MQ) sK[t] = 0.f;
  __syncthreads();
  const int n = blockIdx.x * 256 + t;
  const float* qrow = Qf + ((size_t)b * NPOS + n) * MQ;
  const float* krow = Kf + ((size_t)b * NPOS + n) * MQ;
  const float4* q4 = (const float4*)qrow;
  const float4* k4 = (const float4*)krow;
  float ssq = 0.f, ssk = 0.f;
#pragma unroll
  for (int i = 0; i < 16; ++i) {
    float4 q = q4[i]; ssq += q.x * q.x + q.y * q.y + q.z * q.z + q.w * q.w;
    float4 k = k4[i]; ssk += k.x * k.x + k.y * k.y + k.z * k.z + k.w * k.w;
  }
  const float rq = rsqrtf(ssq), rk = rsqrtf(ssk);
  unsigned short* qn  = Qn + ((size_t)b * NPOS + n) * MQ;
  unsigned short* knb = Kn + (size_t)b * MQ * NPOS + n;
#pragma unroll 8
  for (int m = 0; m < MQ; ++m) {
    qn[m] = f2bf(qrow[m] * rq);
    float kv = krow[m] * rk;
    knb[(size_t)m * NPOS] = f2bf(kv);
    atomicAdd(&sK[m], kv);
  }
  __syncthreads();
  if (t < MQ) atomicAdd(&Ksum[b * MQ + t], sK[t]);
}

// K2b: vsum[b][c] = sum_n Vc[b][c][n].  grid (C, B), block 256
__global__ void k_vsum(const unsigned short* __restrict__ Vc, float* __restrict__ vsum) {
  __shared__ float red[256];
  const int t = threadIdx.x, c = blockIdx.x, b = blockIdx.y;
  const unsigned short* vrow = Vc + ((size_t)b * CH + c) * NPOS;
  float s = 0.f;
#pragma unroll
  for (int i = 0; i < NPOS / 256; ++i) s += bf2f(vrow[t + i * 256]);
  red[t] = s; __syncthreads();
  for (int off = 128; off > 0; off >>= 1) {
    if (t < off) red[t] += red[t + off];
    __syncthreads();
  }
  if (t == 0) vsum[b * CH + c] = red[0];
}

// K3: tailor[b][n] = 1 / (N + sum_m Qn[n][m] * Ksum[b][m]).  grid (16, B)
__global__ void k_tailor(const unsigned short* __restrict__ Qn,
                         const float* __restrict__ Ksum, float* __restrict__ tailor) {
  __shared__ float s[MQ];
  const int t = threadIdx.x, b = blockIdx.y;
  if (t < MQ) s[t] = Ksum[b * MQ + t];
  __syncthreads();
  const int n = blockIdx.x * 256 + t;
  const unsigned short* qn = Qn + ((size_t)b * NPOS + n) * MQ;
  float dot = 0.f;
#pragma unroll 8
  for (int m = 0; m < MQ; ++m) dot += bf2f(qn[m]) * s[m];
  tailor[(size_t)b * NPOS + n] = 1.f / ((float)NPOS + dot);
}

// ---------------------------------------------------------------------------
// K4: matrixT[b][c][m] = sum_n Kn[m][n] * Vc[c][n]   (M=64, cols=512, K=4096)
// 8 waves/block, each wave: 1 m-tile x 2 c-tiles, register-pipelined K loop.
// grid (8, B)
// ---------------------------------------------------------------------------
__global__ void __launch_bounds__(256)
k_gemm_kv(const unsigned short* __restrict__ Kn, const unsigned short* __restrict__ Vc,
          unsigned short* __restrict__ matT) {
  const int b    = blockIdx.y;
  const int w    = threadIdx.x >> 5;
  const int lane = threadIdx.x & 31, l15 = lane & 15, hi = lane >> 4;
  const int id  = blockIdx.x * 8 + w;          // 0..63
  const int mt  = id & 3;                      // 4 m-tiles
  const int ct2 = id >> 2;                     // 16 c-tile pairs

  const unsigned short* Arow  = Kn + (size_t)b * MQ * NPOS + (size_t)(mt * 16 + l15) * NPOS;
  const unsigned short* Bcol0 = Vc + (size_t)b * CH * NPOS + (size_t)(ct2 * 32 + l15) * NPOS;
  const unsigned short* Bcol1 = Bcol0 + (size_t)16 * NPOS;

  v8f acc0 = {}, acc1 = {};
  v16bf af = loadA(Arow + hi * 16);
  v16bf b0 = loadB(Bcol0 + hi * 16);
  v16bf b1 = loadB(Bcol1 + hi * 16);
  for (int kk = 32; kk <= NPOS; kk += 32) {
    v16bf afn, b0n, b1n;
    const bool more = kk < NPOS;
    if (more) {
      __builtin_prefetch(Arow + kk + 512, 0, 1);
      __builtin_prefetch(Bcol0 + kk + 512, 0, 1);
      __builtin_prefetch(Bcol1 + kk + 512, 0, 1);
      afn = loadA(Arow + kk + hi * 16);
      b0n = loadB(Bcol0 + kk + hi * 16);
      b1n = loadB(Bcol1 + kk + hi * 16);
    }
    acc0 = wmma_bf16(af, b0, acc0);
    acc1 = wmma_bf16(af, b1, acc1);
    if (more) { af = afn; b0 = b0n; b1 = b1n; }
  }

  unsigned short* o0 =
      matT + (size_t)b * CH * MQ + (size_t)(ct2 * 32 + l15) * MQ + mt * 16;
  unsigned short* o1 = o0 + (size_t)16 * MQ;
#pragma unroll
  for (int rr = 0; rr < 8; ++rr) {
    o0[rr + 8 * hi] = f2bf(acc0[rr]);
    o1[rr + 8 * hi] = f2bf(acc1[rr]);
  }
}

// ---------------------------------------------------------------------------
// K5: S[n][c] = sum_m Qn[n][m]*matT[c][m]; out = x + gamma*(vsum+S)*tailor
// grid (N/128=32, C/16=32, B), 8 waves/block (each 16n x 16c)
// ---------------------------------------------------------------------------
__global__ void __launch_bounds__(256)
k_gemm_out(const unsigned short* __restrict__ Qn, const unsigned short* __restrict__ matT,
           const float* __restrict__ vsum, const float* __restrict__ tailor,
           const float* __restrict__ x, const float* __restrict__ gamma,
           float* __restrict__ out) {
  const int b    = blockIdx.z;
  const int w    = threadIdx.x >> 5;
  const int lane = threadIdx.x & 31, l15 = lane & 15, hi = lane >> 4;
  const int n0 = blockIdx.x * 128 + w * 16;
  const int c0 = blockIdx.y * 16;

  const unsigned short* Arow = Qn + (size_t)b * NPOS * MQ + (size_t)(n0 + l15) * MQ;
  const unsigned short* Bcol = matT + (size_t)b * CH * MQ + (size_t)(c0 + l15) * MQ;

  // K = 64: both k-step fragments issued up front, then 2 WMMAs.
  v16bf a0 = loadA(Arow + hi * 16);
  v16bf a1 = loadA(Arow + 32 + hi * 16);
  v16bf b0 = loadB(Bcol + hi * 16);
  v16bf b1 = loadB(Bcol + 32 + hi * 16);
  v8f acc = {};
  acc = wmma_bf16(a0, b0, acc);
  acc = wmma_bf16(a1, b1, acc);

  const int c = c0 + l15;
  const float vs = vsum[b * CH + c];
  const float g  = gamma[0];
  const float* xrow = x + ((size_t)b * CH + c) * NPOS;
  float* orow       = out + ((size_t)b * CH + c) * NPOS;
  const float* tl   = tailor + (size_t)b * NPOS;
#pragma unroll
  for (int rr = 0; rr < 8; ++rr) {
    int n = n0 + rr + 8 * hi;
    orow[n] = xrow[n] + g * (vs + acc[rr]) * tl[n];
  }
}

// ---------------------------------------------------------------------------
// Host launcher
// ---------------------------------------------------------------------------
static constexpr size_t al256(size_t v) { return (v + 255) & ~(size_t)255; }

extern "C" void kernel_launch(void* const* d_in, const int* in_sizes, int n_in,
                              void* d_out, int out_size, void* d_ws, size_t ws_size,
                              hipStream_t stream) {
  (void)in_sizes; (void)n_in; (void)out_size; (void)ws_size;
  const float* x     = (const float*)d_in[0];
  const float* Wq    = (const float*)d_in[1];
  const float* bq    = (const float*)d_in[2];
  const float* Wk    = (const float*)d_in[3];
  const float* bk    = (const float*)d_in[4];
  const float* Wv    = (const float*)d_in[5];
  const float* bv    = (const float*)d_in[6];
  const float* gamma = (const float*)d_in[7];
  float* out = (float*)d_out;

  char* ws = (char*)d_ws;
  size_t off = 0;
  auto grab = [&](size_t bytes) { char* p = ws + off; off += al256(bytes); return p; };

  unsigned short* xT   = (unsigned short*)grab((size_t)BATCH * NPOS * CH * 2);
  unsigned short* Wcat = (unsigned short*)grab((size_t)RTOT * CH * 2);
  float*          Qf   = (float*)grab((size_t)BATCH * NPOS * MQ * 4);
  float*          Kf   = (float*)grab((size_t)BATCH * NPOS * MQ * 4);
  unsigned short* Vc   = (unsigned short*)grab((size_t)BATCH * CH * NPOS * 2);
  unsigned short* Qn   = (unsigned short*)grab((size_t)BATCH * NPOS * MQ * 2);
  unsigned short* Kn   = (unsigned short*)grab((size_t)BATCH * MQ * NPOS * 2);
  unsigned short* matT = (unsigned short*)grab((size_t)BATCH * CH * MQ * 2);
  float*          Ksum = (float*)grab((size_t)BATCH * MQ * 4);
  float*          vsum = (float*)grab((size_t)BATCH * CH * 4);
  float*          tlr  = (float*)grab((size_t)BATCH * NPOS * 4);

  k_transpose<<<dim3(NPOS / 32, CH / 32, BATCH), dim3(32, 8), 0, stream>>>(x, xT);
  k_pack_w<<<(RTOT * CH) / 256, 256, 0, stream>>>(Wq, Wk, Wv, Wcat);
  k_init_ksum<<<2, 256, 0, stream>>>(Ksum);
  k_gemm_qkv<<<dim3(NPOS / 128, RTOT / 64, BATCH), 256, 0, stream>>>(
      xT, Wcat, bq, bk, bv, Qf, Kf, Vc);
  k_norm<<<dim3(NPOS / 256, BATCH), 256, 0, stream>>>(Qf, Kf, Qn, Kn, Ksum);
  k_vsum<<<dim3(CH, BATCH), 256, 0, stream>>>(Vc, vsum);
  k_tailor<<<dim3(NPOS / 256, BATCH), 256, 0, stream>>>(Qn, Ksum, tlr);
  k_gemm_kv<<<dim3(8, BATCH), 256, 0, stream>>>(Kn, Vc, matT);
  k_gemm_out<<<dim3(NPOS / 128, CH / 16, BATCH), 256, 0, stream>>>(
      Qn, matT, vsum, tlr, x, gamma, out);
}
